// LambdaRankLoss_73040213835877
// MI455X (gfx1250) — compile-verified, matrix operand order checked by reading, not run on hardware
//
#include <hip/hip_runtime.h>
#include <hip/hip_bf16.h>
#include <math.h>

typedef __attribute__((ext_vector_type(16))) _Float16 v16h;
typedef __attribute__((ext_vector_type(8)))  float    v8f;

#define NUM_QUERIES 2048
#define G 128

// One workgroup (128 threads = 4 wave32) per query group.
__global__ __launch_bounds__(128) void lambdarank_per_query(
    const float* __restrict__ scores,
    const int*   __restrict__ labels,
    float*       __restrict__ partial)   // [NUM_QUERIES][2] = (loss, count)
{
    __shared__ float s_s[G];   // scores
    __shared__ float s_l[G];   // labels (as float; values 0..4 exact)
    __shared__ float s_g[G];   // rank counts, then gains
    __shared__ float red_l[G];
    __shared__ float red_c[G];

    const int q    = blockIdx.x;
    const int t    = threadIdx.x;
    const int base = q * G;

    s_s[t] = scores[base + t];
    s_l[t] = (float)labels[base + t];
    __syncthreads();

    // ---------------------------------------------------------------
    // Rank counting as a boolean matmul on the matrix core:
    //   rank_i = sum_j [ s_j > s_i  ||  (s_j == s_i && j < i) ]
    // A = 16x32 f16 comparison tile, B = all-ones 32x16 f16,
    // chained V_WMMA_F32_16X16X32_F16 over K = 128.
    // ---------------------------------------------------------------
    const int lane = t & 31;
    const int wave = t >> 5;
    const int m    = lane & 15;   // row within 16-row block
    const int half = lane >> 4;   // K-half select per A-operand layout

    v16h bones;
#pragma unroll
    for (int e = 0; e < 16; ++e) bones[e] = (_Float16)1.0f;

#pragma unroll
    for (int rbi = 0; rbi < 2; ++rbi) {
        const int rb = wave * 2 + rbi;      // row block 0..7
        const int i  = rb * 16 + m;         // this lane's matrix row
        const float si = s_s[i];
        v8f acc = {0.f, 0.f, 0.f, 0.f, 0.f, 0.f, 0.f, 0.f};
#pragma unroll
        for (int kc = 0; kc < 4; ++kc) {    // K chunks of 32
            const int k0 = kc * 32;
            v16h a;
#pragma unroll
            for (int e = 0; e < 16; ++e) {
                // 16-bit A 16x32 layout: lanes 0-15: e0-7 -> K0-7, e8-15 -> K16-23
                //                        lanes 16-31: shifted by 8 in K.
                const int klocal = (e < 8) ? (half * 8 + e)
                                           : (16 + half * 8 + (e - 8));
                const int j = k0 + klocal;
                const float sj  = s_s[j];
                const float hit = (sj > si || (sj == si && j < i)) ? 1.0f : 0.0f;
                a[e] = (_Float16)hit;
            }
            acc = __builtin_amdgcn_wmma_f32_16x16x32_f16(
                      false, a, false, bones, (short)0, acc, false, false);
        }
        // C/D layout: lane 0 holds D[M=r][N=0]=cnt[r] in acc[r];
        //             lane 16 holds D[M=8+r][N=0]=cnt[8+r].
        if (lane == 0) {
#pragma unroll
            for (int r = 0; r < 8; ++r) s_g[rb * 16 + r] = acc[r];
        } else if (lane == 16) {
#pragma unroll
            for (int r = 0; r < 8; ++r) s_g[rb * 16 + 8 + r] = acc[r];
        }
    }
    __syncthreads();

    // count -> gain = 1 / log2(rank + 2)   (same-slot rewrite, no hazard)
    {
        const float cnt = s_g[t];
        s_g[t] = 1.0f / log2f(cnt + 2.0f);
    }
    __syncthreads();

    // ---------------------------------------------------------------
    // Pairwise loss. Term is symmetric in (i,j): sum over all j != i,
    // halve at the end. Each thread owns one doc -> perfect balance.
    // ---------------------------------------------------------------
    const float si = s_s[t];
    const float li = s_l[t];
    const float gi = s_g[t];
    float acc_loss = 0.0f;
    float acc_cnt  = 0.0f;
    for (int j = 0; j < G; ++j) {
        const float lj = s_l[j];
        if (lj != li) {
            const float sj  = s_s[j];
            const float gj  = s_g[j];
            const float lam = fabsf(gi - gj);
            const float sgn = (li > lj) ? 1.0f : -1.0f;  // sign(ldiff)
            const float x   = -sgn * (si - sj);          // -sigma*sign(ldiff)*sdiff
            const float sp  = fmaxf(x, 0.0f) + log1pf(__expf(-fabsf(x)));
            acc_loss = fmaf(lam, sp, acc_loss);
            acc_cnt += 1.0f;
        }
    }

    red_l[t] = acc_loss;
    red_c[t] = acc_cnt;
    __syncthreads();
#pragma unroll
    for (int off = 64; off > 0; off >>= 1) {
        if (t < off) {
            red_l[t] += red_l[t + off];
            red_c[t] += red_c[t + off];
        }
        __syncthreads();
    }
    if (t == 0) {
        partial[2 * q]     = 0.5f * red_l[0];  // ordered -> unordered pairs
        partial[2 * q + 1] = 0.5f * red_c[0];  // exact: ordered count is even
    }
}

// Deterministic single-block reduction over per-query partials.
__global__ __launch_bounds__(256) void lambdarank_finalize(
    const float* __restrict__ partial, float* __restrict__ out)
{
    __shared__ float rl[256];
    __shared__ float rc[256];
    float l = 0.0f, c = 0.0f;
    for (int q = (int)threadIdx.x; q < NUM_QUERIES; q += 256) {
        l += partial[2 * q];
        c += partial[2 * q + 1];
    }
    rl[threadIdx.x] = l;
    rc[threadIdx.x] = c;
    __syncthreads();
    for (int off = 128; off > 0; off >>= 1) {
        if ((int)threadIdx.x < off) {
            rl[threadIdx.x] += rl[threadIdx.x + off];
            rc[threadIdx.x] += rc[threadIdx.x + off];
        }
        __syncthreads();
    }
    if (threadIdx.x == 0) out[0] = rl[0] / fmaxf(rc[0], 1.0f);
}

extern "C" void kernel_launch(void* const* d_in, const int* in_sizes, int n_in,
                              void* d_out, int out_size, void* d_ws, size_t ws_size,
                              hipStream_t stream) {
    (void)in_sizes; (void)n_in; (void)out_size; (void)ws_size;
    const float* scores = (const float*)d_in[0];
    const int*   labels = (const int*)d_in[1];
    // d_in[2] (query_ids) unused: groups are contiguous and equal-sized.
    float* partial = (float*)d_ws;          // 2 * NUM_QUERIES floats
    float* out     = (float*)d_out;

    lambdarank_per_query<<<NUM_QUERIES, G, 0, stream>>>(scores, labels, partial);
    lambdarank_finalize<<<1, 256, 0, stream>>>(partial, out);
}
